// Decoder_63900523430071
// MI455X (gfx1250) — compile-verified, matrix operand order checked by reading, not run on hardware
//
#include <hip/hip_runtime.h>
#include <cstdint>
#include <cstddef>

typedef __bf16 bf16;
typedef __attribute__((ext_vector_type(16))) __bf16 v16bf;
typedef __attribute__((ext_vector_type(8)))  float  v8f;

#define B_   32
#define TENC 256
#define TDEC 128
#define NMEL 80
#define ENCD 512
#define DECD 1024
#define PRED 256
#define ARNN 1024
#define ATTD 128
#define LOCF 32
#define LOCK 31

// ---------------------------------------------------------------------------
// Packed bf16 tile layout (ISA 7.12.2, 16-bit A-matrix 16x32):
//   element e in [0,512): lane = e>>4, j = (e>>1)&7, h = e&1
//   row (M or N) = lane & 15
//   k = 8*(lane>>4) + 16*(j>>2) + 2*(j&3) + h
// Tile (rt, kt) of a [rows x K] matrix stored at ((rt*KtTot)+kt)*512 + e.
// ---------------------------------------------------------------------------

__device__ __forceinline__ void packStore(bf16* __restrict__ P, int KtTot, int ktOff,
                                          int m, int k, float v) {
  int kg = (ktOff << 5) + k;
  int ktp = kg >> 5, kk = kg & 31;
  int mt = m >> 4, mm = m & 15;
  int lp = mm + (((kk >> 3) & 1) << 4);
  int j  = ((kk >> 4) << 2) + ((kk & 7) >> 1);
  P[((size_t)(mt * KtTot + ktp) << 9) + (lp << 4) + (j << 1) + (kk & 1)] = (bf16)v;
}

// Pack a row-major f32 matrix piece [rows x Kvalid] (ld = leading dim) into
// WMMA tile layout, tiles (nt, kt+ktOff) with total KtTot k-tiles per row-tile.
__global__ void pack_mat(const float* __restrict__ S, bf16* __restrict__ P,
                         int rows, int ld, int Kvalid,
                         int Ntiles, int KtP, int ktOff, int KtTot) {
  int idx = blockIdx.x * blockDim.x + threadIdx.x;
  int total = Ntiles * KtP * 512;
  if (idx >= total) return;
  int e  = idx & 511;
  int kt = (idx >> 9) % KtP;
  int nt = (idx >> 9) / KtP;
  int lp = e >> 4, j = (e >> 1) & 7, h = e & 1;
  int n  = (nt << 4) + (lp & 15);
  int kk = ((lp >> 4) << 3) + ((j >> 2) << 4) + ((j & 3) << 1) + h;
  int k  = (kt << 5) + kk;
  float v = (n < rows && k < Kvalid) ? S[(size_t)n * ld + k] : 0.f;
  P[((size_t)(nt * KtTot + kt + ktOff) << 9) + e] = (bf16)v;
}

// Teacher-forcing input frame -> packed A tiles [32 x 96] (K padded 80->96).
__global__ void frame_pack(const float* __restrict__ din, bf16* __restrict__ xp, int t) {
  int idx = blockIdx.x * blockDim.x + threadIdx.x;
  if (idx >= 2 * 3 * 512) return;
  int e  = idx & 511;
  int kt = (idx >> 9) % 3;
  int mt = (idx >> 9) / 3;
  int lp = e >> 4, j = (e >> 1) & 7, h = e & 1;
  int m  = (mt << 4) + (lp & 15);
  int kk = ((lp >> 4) << 3) + ((j >> 2) << 4) + ((j & 3) << 1) + h;
  int k  = (kt << 5) + kk;
  float v = (t > 0 && k < NMEL) ? din[(size_t)m * NMEL * TDEC + (size_t)k * TDEC + (t - 1)] : 0.f;
  xp[((size_t)(mt * 3 + kt) << 9) + e] = (bf16)v;
}

// Per-tile GEMM epilogue (inlined; flags tested at run time, uniform per launch)
__device__ __forceinline__ void gemmEpilogue(v8f acc, int mt, int nt, int lane,
                                             const float* __restrict__ bias,
                                             float* __restrict__ Cf, int ldc,
                                             bf16* __restrict__ Cp, int ktOff, int KtTotP,
                                             float* __restrict__ outMel,
                                             float* __restrict__ outGate,
                                             int tstep, int flags) {
  int nn = lane & 15, mh = lane >> 4;
  int n = (nt << 4) + nn;
  float bs = (flags & 8) ? bias[n] : 0.f;
#pragma unroll
  for (int r = 0; r < 8; ++r) {
    int m = (mt << 4) + r + (mh << 3);
    float v = acc[r] + bs;
    if (flags & 1) v = fmaxf(v, 0.f);
    if (flags & 2) Cf[(size_t)m * ldc + n] = v;
    if (flags & 4) packStore(Cp, KtTotP, ktOff, m, n, v);
    if (flags & 16) {
      if (n < NMEL)       outMel[(size_t)m * NMEL * TDEC + (size_t)n * TDEC + tstep] = v;
      else if (n == NMEL) outGate[m * TDEC + tstep] = v;
    }
  }
}

// ---------------------------------------------------------------------------
// Generic WMMA GEMM, register-blocked 4 N-tiles per wave (one A tile feeds 4
// WMMAs -> 4x less redundant activation traffic from L2). Nt must be a
// multiple of 4 (host pads). All accumulators are explicit scalars: no
// dynamically-indexed register arrays (avoids v_movrels indirect addressing).
// flags: 1=relu, 2=write f32 C, 4=write packed bf16 C, 8=add bias, 16=proj out
// ---------------------------------------------------------------------------
__global__ void gemm_bf16_wmma(const bf16* __restrict__ A, const bf16* __restrict__ W,
                               const float* __restrict__ bias,
                               int Mt, int Nt, int Kt,
                               float* __restrict__ Cf, int ldc,
                               bf16* __restrict__ Cp, int ktOff, int KtTotP,
                               float* __restrict__ outMel, float* __restrict__ outGate,
                               int tstep, int flags) {
  int lane = threadIdx.x & 31;
  int wave = (blockIdx.x * blockDim.x + threadIdx.x) >> 5;
  int nw   = (gridDim.x * blockDim.x) >> 5;
  int ntg  = Nt >> 2;                  // N-tile groups of 4 (Nt % 4 == 0)
  int groups = Mt * ntg;
  for (int g = wave; g < groups; g += nw) {
    int mt  = g / ntg;
    int nt0 = (g - mt * ntg) << 2;

    v8f a0 = {}, a1 = {}, a2 = {}, a3 = {};
    const v16bf* aP = (const v16bf*)(A + (((size_t)mt * Kt) << 9) + (lane << 4));
    size_t wBase = ((size_t)nt0 * Kt) << 9;
    const v16bf* w0 = (const v16bf*)(W + wBase + (lane << 4));
    const v16bf* w1 = (const v16bf*)(W + wBase + (((size_t)Kt) << 9)     + (lane << 4));
    const v16bf* w2 = (const v16bf*)(W + wBase + (((size_t)Kt) << 9) * 2 + (lane << 4));
    const v16bf* w3 = (const v16bf*)(W + wBase + (((size_t)Kt) << 9) * 3 + (lane << 4));

    for (int kt = 0; kt < Kt; ++kt) {
      v16bf av = aP[kt * 32];
      a0 = __builtin_amdgcn_wmma_f32_16x16x32_bf16(false, av, false, w0[kt * 32],
                                                   (short)0, a0, false, false);
      a1 = __builtin_amdgcn_wmma_f32_16x16x32_bf16(false, av, false, w1[kt * 32],
                                                   (short)0, a1, false, false);
      a2 = __builtin_amdgcn_wmma_f32_16x16x32_bf16(false, av, false, w2[kt * 32],
                                                   (short)0, a2, false, false);
      a3 = __builtin_amdgcn_wmma_f32_16x16x32_bf16(false, av, false, w3[kt * 32],
                                                   (short)0, a3, false, false);
    }

    gemmEpilogue(a0, mt, nt0 + 0, lane, bias, Cf, ldc, Cp, ktOff, KtTotP, outMel, outGate, tstep, flags);
    gemmEpilogue(a1, mt, nt0 + 1, lane, bias, Cf, ldc, Cp, ktOff, KtTotP, outMel, outGate, tstep, flags);
    gemmEpilogue(a2, mt, nt0 + 2, lane, bias, Cf, ldc, Cp, ktOff, KtTotP, outMel, outGate, tstep, flags);
    gemmEpilogue(a3, mt, nt0 + 3, lane, bias, Cf, ldc, Cp, ktOff, KtTotP, outMel, outGate, tstep, flags);
  }
}

// ---------------------------------------------------------------------------
// LSTM pointwise: gates [32 x 4H] (i,f,g,o) -> c,h; scatter h into up to 3
// packed activation buffers.
// ---------------------------------------------------------------------------
__global__ void lstm_point(const float* __restrict__ gates,
                           float* __restrict__ c, float* __restrict__ h, int H,
                           bf16* p1, int ko1, int kt1,
                           bf16* p2, int ko2, int kt2,
                           bf16* p3, int ko3, int kt3) {
  int idx = blockIdx.x * blockDim.x + threadIdx.x;
  if (idx >= B_ * H) return;
  int b = idx / H, j = idx - b * H;
  const float* g = gates + (size_t)b * 4 * H;
  float gi = g[j], gf = g[H + j], gg = g[2 * H + j], go = g[3 * H + j];
  float si = 1.f / (1.f + __expf(-gi));
  float sf = 1.f / (1.f + __expf(-gf));
  float so = 1.f / (1.f + __expf(-go));
  float cn = sf * c[idx] + si * tanhf(gg);
  float hn = so * tanhf(cn);
  c[idx] = cn; h[idx] = hn;
  if (p1) packStore(p1, kt1, ko1, b, j, hn);
  if (p2) packStore(p2, kt2, ko2, b, j, hn);
  if (p3) packStore(p3, kt3, ko3, b, j, hn);
}

// ---------------------------------------------------------------------------
// Location-sensitive attention: conv(aw_prev) -> energies -> softmax -> ctx.
// One block per batch row, 256 threads (one per encoder timestep).
// ---------------------------------------------------------------------------
__global__ void att_kernel(const float* __restrict__ convsum,   // [32 x 31]
                           const float* __restrict__ loc_w,     // [128 x 32]
                           const float* __restrict__ att_v,     // [128]
                           const float* __restrict__ query,     // [32 x 128]
                           const float* __restrict__ pm,        // [32*256 x 128]
                           const float* __restrict__ memory,    // [32 x 256 x 512]
                           const int*   __restrict__ mlen,      // [32]
                           float* __restrict__ awprev,          // [32 x 256] in/out
                           float* __restrict__ alignOut,        // [32 x 128 x 256]
                           float* __restrict__ ctx,             // [32 x 512]
                           bf16* __restrict__ attinP,
                           bf16* __restrict__ decinP,
                           bf16* __restrict__ projinP, int tstep) {
  int b = blockIdx.x;
  int t = threadIdx.x;
  __shared__ float sPrev[TENC];
  __shared__ float sRed[TENC];
  __shared__ float sAw[TENC];

  sPrev[t] = awprev[b * TENC + t];
  __syncthreads();

  // location conv (both channels carry the same aw -> channel-summed weights)
  float conv[LOCF];
#pragma unroll
  for (int f = 0; f < LOCF; ++f) conv[f] = 0.f;
  for (int k = 0; k < LOCK; ++k) {
    int tt = t + k - (LOCK - 1) / 2;
    if (tt >= 0 && tt < TENC) {
      float x = sPrev[tt];
      const float* cw = convsum + k;
#pragma unroll
      for (int f = 0; f < LOCF; ++f) conv[f] += cw[f * LOCK] * x;
    }
  }

  // energies
  const float* pmRow = pm + ((size_t)b * TENC + t) * ATTD;
  const float* qRow  = query + b * ATTD;
  float e = 0.f;
  for (int d = 0; d < ATTD; ++d) {
    float lf = 0.f;
    const float* lw = loc_w + d * LOCF;
#pragma unroll
    for (int f = 0; f < LOCF; ++f) lf += conv[f] * lw[f];
    e += att_v[d] * tanhf(qRow[d] + pmRow[d] + lf);
  }
  bool valid = t < mlen[b];
  float ev = valid ? e : -INFINITY;

  // softmax over t (block reduce)
  sRed[t] = ev; __syncthreads();
  for (int s = 128; s > 0; s >>= 1) {
    if (t < s) sRed[t] = fmaxf(sRed[t], sRed[t + s]);
    __syncthreads();
  }
  float vmax = sRed[0]; __syncthreads();
  float ex = valid ? __expf(ev - vmax) : 0.f;
  sRed[t] = ex; __syncthreads();
  for (int s = 128; s > 0; s >>= 1) {
    if (t < s) sRed[t] += sRed[t + s];
    __syncthreads();
  }
  float aw = ex / sRed[0];
  sAw[t] = aw;
  alignOut[(size_t)b * TDEC * TENC + (size_t)tstep * TENC + t] = aw;
  awprev[b * TENC + t] = aw;
  __syncthreads();

  // ctx = sum_t aw[t] * memory[b,t,:]
  for (int d = t; d < ENCD; d += TENC) {
    const float* mb = memory + ((size_t)b * TENC) * ENCD + d;
    float s = 0.f;
    for (int tt = 0; tt < TENC; ++tt) s += sAw[tt] * mb[(size_t)tt * ENCD];
    ctx[b * ENCD + d] = s;
    packStore(attinP, 56, 8,  b, d, s);  // att-LSTM input ctx region (next step)
    packStore(decinP, 80, 32, b, d, s);  // dec-LSTM input ctx region (this step)
    packStore(projinP, 48, 32, b, d, s); // proj input ctx region (this step)
  }
}

// one-time prep: bias sums, channel-summed conv, proj+gate weight concat
__global__ void prep_kernel(const float* __restrict__ bih_a, const float* __restrict__ bhh_a,
                            const float* __restrict__ bih_d, const float* __restrict__ bhh_d,
                            const float* __restrict__ loc_conv,
                            const float* __restrict__ proj_w, const float* __restrict__ proj_b,
                            const float* __restrict__ gate_w, const float* __restrict__ gate_b,
                            float* __restrict__ bias_att, float* __restrict__ bias_dec,
                            float* __restrict__ bias_proj, float* __restrict__ convsum,
                            float* __restrict__ Wcomb) {
  int i = blockIdx.x * blockDim.x + threadIdx.x;
  if (i < 4096) { bias_att[i] = bih_a[i] + bhh_a[i]; bias_dec[i] = bih_d[i] + bhh_d[i]; }
  if (i < 128)  bias_proj[i] = (i < NMEL) ? proj_b[i] : ((i == NMEL) ? gate_b[0] : 0.f);
  if (i < LOCF * LOCK) {
    int f = i / LOCK, k = i - f * LOCK;
    convsum[i] = loc_conv[f * 2 * LOCK + k] + loc_conv[f * 2 * LOCK + LOCK + k];
  }
  if (i < 96 * 1536) {
    int n = i / 1536, k = i - n * 1536;
    Wcomb[i] = (n < NMEL) ? proj_w[n * 1536 + k] : ((n == NMEL) ? gate_w[k] : 0.f);
  }
}

__global__ void zero_kernel(uint32_t* __restrict__ p, size_t n) {
  size_t i = (size_t)blockIdx.x * blockDim.x + threadIdx.x;
  size_t stride = (size_t)gridDim.x * blockDim.x;
  for (; i < n; i += stride) p[i] = 0u;
}

// ---------------------------------------------------------------------------
extern "C" void kernel_launch(void* const* d_in, const int* in_sizes, int n_in,
                              void* d_out, int out_size, void* d_ws, size_t ws_size,
                              hipStream_t stream) {
  (void)in_sizes; (void)n_in; (void)out_size; (void)ws_size;

  const float* memory   = (const float*)d_in[0];
  const float* dinputs  = (const float*)d_in[1];
  const int*   mlen     = (const int*)d_in[2];
  const float* pre_w1   = (const float*)d_in[3];
  const float* pre_b1   = (const float*)d_in[4];
  const float* pre_w2   = (const float*)d_in[5];
  const float* pre_b2   = (const float*)d_in[6];
  const float* arnn_wih = (const float*)d_in[7];
  const float* arnn_whh = (const float*)d_in[8];
  const float* arnn_bih = (const float*)d_in[9];
  const float* arnn_bhh = (const float*)d_in[10];
  const float* att_wq   = (const float*)d_in[11];
  const float* att_wm   = (const float*)d_in[12];
  const float* att_v    = (const float*)d_in[13];
  const float* loc_conv = (const float*)d_in[14];
  const float* loc_w    = (const float*)d_in[15];
  const float* drnn_wih = (const float*)d_in[16];
  const float* drnn_whh = (const float*)d_in[17];
  const float* drnn_bih = (const float*)d_in[18];
  const float* drnn_bhh = (const float*)d_in[19];
  const float* proj_w   = (const float*)d_in[20];
  const float* proj_b   = (const float*)d_in[21];
  const float* gate_w   = (const float*)d_in[22];
  const float* gate_b   = (const float*)d_in[23];

  float* outMel   = (float*)d_out;                        // [32,80,128]
  float* outGate  = outMel + (size_t)B_ * NMEL * TDEC;    // [32,128]
  float* outAlign = outGate + (size_t)B_ * TDEC;          // [32,128,256]

  // ---- workspace layout ----
  char* ws = (char*)d_ws;
  size_t off = 0;
  auto alloc = [&](size_t bytes) { size_t o = off; off = (off + bytes + 255) & ~(size_t)255; return o; };

  float* pm        = (float*)(ws + alloc((size_t)B_ * TENC * ATTD * 4));
  float* gates_att = (float*)(ws + alloc((size_t)B_ * 4096 * 4));
  float* gates_dec = (float*)(ws + alloc((size_t)B_ * 4096 * 4));
  float* query     = (float*)(ws + alloc((size_t)B_ * ATTD * 4));
  float* bias_att  = (float*)(ws + alloc(4096 * 4));
  float* bias_dec  = (float*)(ws + alloc(4096 * 4));
  float* bias_proj = (float*)(ws + alloc(128 * 4));
  float* convsum   = (float*)(ws + alloc(LOCF * LOCK * 4));
  float* Wcomb     = (float*)(ws + alloc(96 * 1536 * 4));

  size_t zBegin = off;                                    // zero-init region start
  float* att_h   = (float*)(ws + alloc((size_t)B_ * ARNN * 4));
  float* att_c   = (float*)(ws + alloc((size_t)B_ * ARNN * 4));
  float* dec_h   = (float*)(ws + alloc((size_t)B_ * DECD * 4));
  float* dec_c   = (float*)(ws + alloc((size_t)B_ * DECD * 4));
  float* ctx     = (float*)(ws + alloc((size_t)B_ * ENCD * 4));
  float* awprev  = (float*)(ws + alloc((size_t)B_ * TENC * 4));
  bf16* xpack    = (bf16*)(ws + alloc((size_t)2 * 3 * 512 * 2));
  bf16* pr1pack  = (bf16*)(ws + alloc((size_t)2 * 8 * 512 * 2));
  bf16* attin    = (bf16*)(ws + alloc((size_t)2 * 56 * 512 * 2));
  bf16* queryA   = (bf16*)(ws + alloc((size_t)2 * 32 * 512 * 2));
  bf16* decin    = (bf16*)(ws + alloc((size_t)2 * 80 * 512 * 2));
  bf16* projin   = (bf16*)(ws + alloc((size_t)2 * 48 * 512 * 2));
  size_t zEnd = off;

  bf16* prew1p = (bf16*)(ws + alloc((size_t)16 * 3 * 512 * 2));
  bf16* prew2p = (bf16*)(ws + alloc((size_t)16 * 8 * 512 * 2));
  bf16* wattp  = (bf16*)(ws + alloc((size_t)256 * 56 * 512 * 2));
  bf16* attwqp = (bf16*)(ws + alloc((size_t)8 * 32 * 512 * 2));
  bf16* wdecp  = (bf16*)(ws + alloc((size_t)256 * 80 * 512 * 2));
  bf16* wprojp = (bf16*)(ws + alloc((size_t)8 * 48 * 512 * 2));   // Nt padded 6->8
  bf16* attwmp = (bf16*)(ws + alloc((size_t)8 * 16 * 512 * 2));
  bf16* memAp  = (bf16*)(ws + alloc((size_t)512 * 16 * 512 * 2));

  auto blocksFor = [](size_t n) { return (unsigned)((n + 255) / 256); };

  // ---- one-time setup ----
  zero_kernel<<<512, 256, 0, stream>>>((uint32_t*)(ws + zBegin), (zEnd - zBegin) / 4);
  prep_kernel<<<blocksFor(96 * 1536), 256, 0, stream>>>(
      arnn_bih, arnn_bhh, drnn_bih, drnn_bhh, loc_conv,
      proj_w, proj_b, gate_w, gate_b,
      bias_att, bias_dec, bias_proj, convsum, Wcomb);

  auto pack = [&](const float* S, bf16* P, int rows, int ld, int Kv,
                  int Ntiles, int KtP, int ktOff, int KtTot) {
    size_t total = (size_t)Ntiles * KtP * 512;
    pack_mat<<<blocksFor(total), 256, 0, stream>>>(S, P, rows, ld, Kv, Ntiles, KtP, ktOff, KtTot);
  };
  pack(pre_w1,   prew1p, PRED, NMEL, NMEL, 16, 3, 0, 3);
  pack(pre_w2,   prew2p, PRED, PRED, PRED, 16, 8, 0, 8);
  pack(arnn_wih, wattp,  4096, 768,  768,  256, 24, 0,  56);
  pack(arnn_whh, wattp,  4096, 1024, 1024, 256, 32, 24, 56);
  pack(att_wq,   attwqp, ATTD, ARNN, ARNN, 8, 32, 0, 32);
  pack(drnn_wih, wdecp,  4096, 1536, 1536, 256, 48, 0,  80);
  pack(drnn_whh, wdecp,  4096, 1024, 1024, 256, 32, 48, 80);
  pack(Wcomb,    wprojp, 96,   1536, 1536, 8, 48, 0, 48);         // rows 96..127 -> 0
  pack(att_wm,   attwmp, ATTD, ENCD, ENCD, 8, 16, 0, 16);
  pack(memory,   memAp,  B_ * TENC, ENCD, ENCD, 512, 16, 0, 16);

  auto gemm = [&](const bf16* A, const bf16* W, const float* bias, int Mt, int Nt, int Kt,
                  float* Cf, int ldc, bf16* Cp, int ktOff, int KtTotP,
                  float* oM, float* oG, int t, int flags) {
    int groups = Mt * (Nt / 4);
    unsigned blocks = (unsigned)((groups + 7) / 8);
    gemm_bf16_wmma<<<blocks, 256, 0, stream>>>(A, W, bias, Mt, Nt, Kt, Cf, ldc,
                                               Cp, ktOff, KtTotP, oM, oG, t, flags);
  };

  // processed_memory = memory @ att_wm.T  : [8192 x 128]
  gemm(memAp, attwmp, nullptr, 512, 8, 16, pm, ATTD, nullptr, 0, 0, nullptr, nullptr, 0, 2);

  // ---- decoder scan ----
  for (int t = 0; t < TDEC; ++t) {
    frame_pack<<<blocksFor(2 * 3 * 512), 256, 0, stream>>>(dinputs, xpack, t);

    // prenet layer 1: relu, pack for layer 2
    gemm(xpack, prew1p, pre_b1, 2, 16, 3, nullptr, 0, pr1pack, 0, 8,
         nullptr, nullptr, 0, 1 | 4 | 8);
    // prenet layer 2: relu, pack into att-LSTM input (kt 0..7 of 56)
    gemm(pr1pack, prew2p, pre_b2, 2, 16, 8, nullptr, 0, attin, 0, 56,
         nullptr, nullptr, 0, 1 | 4 | 8);

    // attention LSTM gates: [pr, ctx_prev, att_h_prev] x [Wih|Whh]
    gemm(attin, wattp, bias_att, 2, 256, 56, gates_att, 4096, nullptr, 0, 0,
         nullptr, nullptr, 0, 2 | 8);
    lstm_point<<<blocksFor((size_t)B_ * ARNN), 256, 0, stream>>>(
        gates_att, att_c, att_h, ARNN,
        attin, 24, 56,      // recurrent input for next step
        decin, 0, 80,       // dec-LSTM input (this step)
        queryA, 0, 32);     // query GEMM input (this step)

    // query = att_h @ att_wq.T
    gemm(queryA, attwqp, nullptr, 2, 8, 32, query, ATTD, nullptr, 0, 0,
         nullptr, nullptr, 0, 2);

    att_kernel<<<B_, TENC, 0, stream>>>(convsum, loc_w, att_v, query, pm,
                                        memory, mlen, awprev, outAlign, ctx,
                                        attin, decin, projin, t);

    // decoder LSTM gates: [att_h, ctx, dec_h_prev] x [Wih|Whh]
    gemm(decin, wdecp, bias_dec, 2, 256, 80, gates_dec, 4096, nullptr, 0, 0,
         nullptr, nullptr, 0, 2 | 8);
    lstm_point<<<blocksFor((size_t)B_ * DECD), 256, 0, stream>>>(
        gates_dec, dec_c, dec_h, DECD,
        decin, 48, 80,      // recurrent input for next step
        projin, 0, 48,      // proj input (this step)
        nullptr, 0, 0);

    // mel/gate projection (N padded to 128), fused scatter into d_out
    gemm(projin, wprojp, bias_proj, 2, 8, 48, nullptr, 0, nullptr, 0, 0,
         outMel, outGate, t, 8 | 16);
  }
}
